// GIKT_22531398435085
// MI455X (gfx1250) — compile-verified, matrix operand order checked by reading, not run on hardware
//
#include <hip/hip_runtime.h>
#include <hip/hip_bf16.h>
#include <math.h>

#define BB  128
#define TTT 199
#define EE  256
#define HH  256
#define KK  8
#define BT  (BB*TTT)     // 25472 = 398*64
#define G4H (4*HH)       // 1024
#define FAW 768          // 3*E

typedef __attribute__((ext_vector_type(16))) __bf16 v16bf;
typedef __attribute__((ext_vector_type(8)))  float  v8f;

// ---------------------------------------------------------------------------
// CDNA5 async global->LDS copy (per-lane 16B), tracked by ASYNCcnt.
// ---------------------------------------------------------------------------
__device__ __forceinline__ void async_copy_b128(unsigned lds_off, const void* gptr) {
  asm volatile("global_load_async_to_lds_b128 %0, %1, off"
               :: "v"(lds_off), "v"((unsigned long long)(uintptr_t)gptr)
               : "memory");
}
__device__ __forceinline__ void wait_async0() {
  asm volatile("s_wait_asynccnt 0x0" ::: "memory");
}

// ---------------------------------------------------------------------------
// bf16 conversion / packing kernels (run once; removes converts from GEMMs)
// ---------------------------------------------------------------------------
__global__ void cvt_bf16(const float* __restrict__ s, __bf16* __restrict__ d, int n) {
  const int i = (blockIdx.x * blockDim.x + threadIdx.x) * 8;
  if (i >= n) return;
  float4 x = *(const float4*)(s + i);
  float4 y = *(const float4*)(s + i + 4);
  union { __bf16 h[8]; uint4 u; } r;
  r.h[0] = (__bf16)x.x; r.h[1] = (__bf16)x.y; r.h[2] = (__bf16)x.z; r.h[3] = (__bf16)x.w;
  r.h[4] = (__bf16)y.x; r.h[5] = (__bf16)y.y; r.h[6] = (__bf16)y.z; r.h[7] = (__bf16)y.w;
  *(uint4*)(d + i) = r.u;
}

// fa = concat([q, a, s], last) -> bf16 (B*T, 768). Each 8-elem chunk stays in one source.
__global__ void pack_fa(const float* __restrict__ q, const float* __restrict__ a,
                        const float* __restrict__ s, __bf16* __restrict__ d) {
  const int i = (blockIdx.x * blockDim.x + threadIdx.x) * 8;
  if (i >= BT * FAW) return;
  const int row = i / FAW;
  const int col = i - row * FAW;
  const int sel = col >> 8;
  const float* src = (sel == 0) ? q : ((sel == 1) ? a : s);
  const float* p = src + (size_t)row * EE + (col & 255);
  float4 x = *(const float4*)(p);
  float4 y = *(const float4*)(p + 4);
  union { __bf16 h[8]; uint4 u; } r;
  r.h[0] = (__bf16)x.x; r.h[1] = (__bf16)x.y; r.h[2] = (__bf16)x.z; r.h[3] = (__bf16)x.w;
  r.h[4] = (__bf16)y.x; r.h[5] = (__bf16)y.y; r.h[6] = (__bf16)y.z; r.h[7] = (__bf16)y.w;
  *(uint4*)(d + i) = r.u;
}

// ---------------------------------------------------------------------------
// LDS-staged bf16 WMMA GEMM: C(MxN) = A(MxK) * W(NxK)^T + addSrc + bias.
// 64x64 macro-tile / 256-thread block (8 waves); wave = 1 A-frag x 2 B-frags
// -> 2 WMMAs per 32-deep k-step. K slices (64x32 bf16 A + B, 8KB) are staged
// into a 2-deep LDS double buffer with async global->LDS b128 copies.
// ACT: 1=ReLU. OUTMODE: 0=row-major; 1=rows are b*T+t, store (t*B+b)*N+col.
// OUTBF: 1=store bf16 output, else f32.
// ---------------------------------------------------------------------------
template<int ACT, int OUTMODE, int OUTBF>
__global__ void gemm_bf16_lds(const __bf16* __restrict__ A, const __bf16* __restrict__ W,
                              const float* __restrict__ bias, const float* __restrict__ addSrc,
                              void* __restrict__ Cout, int M, int N, int Kd,
                              int T, int Bdim) {
  __shared__ char lds[2][8192];   // per buffer: A tile [0,4096), B tile [4096,8192)
  const int tid  = threadIdx.x;
  const int lane = tid & 31;
  const int wv   = tid >> 5;
  const int nbn  = N >> 6;
  const int bm   = blockIdx.x / nbn;
  const int bn   = blockIdx.x - bm * nbn;

  // --- copy role: thread -> (row 0..63, 16B quarter 0..3) of each tile
  const int crow = tid >> 2;
  const int cq   = tid & 3;
  const __bf16* aSrc = A + (size_t)(bm * 64 + crow) * Kd + cq * 8;
  const __bf16* bSrc = W + (size_t)(bn * 64 + crow) * Kd + cq * 8;
  const unsigned ldsBase = (unsigned)(uintptr_t)&lds[0][0];
  const unsigned copyOff = (unsigned)(crow * 64 + cq * 16);

  // --- compute role: wave -> M-tile (wv&3), N-tile pair ((wv>>2)*2 + {0,1})
  const int tm   = wv & 3;
  const int tnp  = wv >> 2;
  const int l16  = lane & 15;
  const int half = lane >> 4;
  const int aoff  = (tm * 16 + l16) * 64 + half * 16;
  const int boff0 = 4096 + (tnp * 32 + l16) * 64 + half * 16;
  const int boff1 = boff0 + 16 * 64;

  auto issue = [&](int kt, int buf) {
    const unsigned la = ldsBase + (unsigned)buf * 8192u + copyOff;
    async_copy_b128(la,        aSrc + (size_t)kt * 32);
    async_copy_b128(la + 4096, bSrc + (size_t)kt * 32);
  };

  union Frag { uint4 u[2]; v16bf v; };
  v8f acc0 = {}, acc1 = {};
  const int nk = Kd >> 5;

  issue(0, 0);
  for (int kt = 0; kt < nk; ++kt) {
    const int cur = kt & 1;
    wait_async0();
    __syncthreads();                  // buf[cur] ready; buf[cur^1] fully consumed
    if (kt + 1 < nk) issue(kt + 1, cur ^ 1);
    const char* Lb = lds[cur];
    Frag fa_, fb0, fb1;
    fa_.u[0] = *(const uint4*)(Lb + aoff);
    fa_.u[1] = *(const uint4*)(Lb + aoff + 32);
    fb0.u[0] = *(const uint4*)(Lb + boff0);
    fb0.u[1] = *(const uint4*)(Lb + boff0 + 32);
    fb1.u[0] = *(const uint4*)(Lb + boff1);
    fb1.u[1] = *(const uint4*)(Lb + boff1 + 32);
    acc0 = __builtin_amdgcn_wmma_f32_16x16x32_bf16(false, fa_.v, false, fb0.v,
                                                   (short)0, acc0, false, false);
    acc1 = __builtin_amdgcn_wmma_f32_16x16x32_bf16(false, fa_.v, false, fb1.v,
                                                   (short)0, acc1, false, false);
  }

  // --- epilogue
  const int colBase = bn * 64 + tnp * 32 + l16;
#pragma unroll
  for (int i = 0; i < 8; ++i) {
    const int row = bm * 64 + tm * 16 + half * 8 + i;   // D frag: VGPR i -> M = i + 8*half
    size_t orow;
    if (OUTMODE == 0) {
      orow = (size_t)row * N;
    } else {
      const int bb = row / T;
      const int tt = row - bb * T;
      orow = ((size_t)tt * Bdim + bb) * N;
    }
#pragma unroll
    for (int t2 = 0; t2 < 2; ++t2) {
      const int col = colBase + t2 * 16;
      float v = (t2 == 0 ? acc0[i] : acc1[i]);
      if (bias)   v += bias[col];
      if (addSrc) v += addSrc[(size_t)row * N + col];
      if (ACT == 1) v = fmaxf(v, 0.f);
      if (OUTBF) ((__bf16*)Cout)[orow + col] = (__bf16)v;
      else       ((float*)Cout)[orow + col] = v;
    }
  }
}

// ---------------------------------------------------------------------------
// LSTM pointwise + init
// ---------------------------------------------------------------------------
__global__ void zero_hc(__bf16* __restrict__ hb, float* __restrict__ c) {
  const int i = blockIdx.x * blockDim.x + threadIdx.x;
  if (i < BB * HH) { hb[i] = (__bf16)0.f; c[i] = 0.f; }
}

__global__ void lstm_pointwise(const float* __restrict__ gates, __bf16* __restrict__ hb,
                               float* __restrict__ c, float* __restrict__ outser, int t) {
  const int idx = blockIdx.x * blockDim.x + threadIdx.x;
  if (idx >= BB * HH) return;
  const int b  = idx >> 8;
  const int hh = idx & (HH - 1);
  const float* g = gates + (size_t)b * G4H;
  const float gi = g[hh], gf = g[HH + hh], gg = g[2 * HH + hh], go = g[3 * HH + hh];
  const float si = 1.f / (1.f + expf(-gi));
  const float sf = 1.f / (1.f + expf(-gf));
  const float so = 1.f / (1.f + expf(-go));
  const float cv = sf * c[idx] + si * tanhf(gg);
  const float hv = so * tanhf(cv);
  c[idx]  = cv;
  hb[idx] = (__bf16)hv;                                  // bf16 h for next step's GEMM
  outser[((size_t)b * TTT + t) * HH + hh] = hv;          // fp32 for attention
}

// ---------------------------------------------------------------------------
// Attention epilogue: one wave per (b,t)
// ---------------------------------------------------------------------------
__device__ __forceinline__ float wred32(float v) {
#pragma unroll
  for (int m = 16; m > 0; m >>= 1) v += __shfl_xor(v, m, 32);
  return v;
}

__global__ void attn_out(const float* __restrict__ outser, const float* __restrict__ ntr,
                         const int* __restrict__ idx,
                         const float* __restrict__ att1_w, const float* __restrict__ att1_b,
                         const float* __restrict__ att2_w, const float* __restrict__ att2_b,
                         float* __restrict__ out) {
  const int wave = threadIdx.x >> 5;
  const int lane = threadIdx.x & 31;
  const int bt = blockIdx.x * (blockDim.x >> 5) + wave;
  if (bt >= BT) return;
  const int b = bt / TTT;
  float nn[8], a1[8];
  const float* nnp = ntr + (size_t)bt * HH + lane * 8;
  const float* a1p = att1_w + lane * 8;
  const float* a2p = att2_w + lane * 8;
  float p2 = 0.f;
#pragma unroll
  for (int i = 0; i < 8; ++i) { nn[i] = nnp[i]; a1[i] = a1p[i]; p2 += nn[i] * a2p[i]; }
  const float f2 = wred32(p2) + att2_b[0];
  float logit[9], f1[9];
#pragma unroll
  for (int j = 0; j < 9; ++j) {
    const float* row;
    if (j == 0) {
      row = outser + (size_t)bt * HH;
    } else {
      const int tt = idx[(size_t)bt * KK + (j - 1)];
      row = outser + ((size_t)b * TTT + tt) * HH;
    }
    row += lane * 8;
    float pl = 0.f, p1 = 0.f;
#pragma unroll
    for (int i = 0; i < 8; ++i) { const float v = row[i]; pl += v * nn[i]; p1 += v * a1[i]; }
    logit[j] = wred32(pl);
    f1[j]    = wred32(p1) + att1_b[0];
  }
  float s[9], mx = -1e30f;
#pragma unroll
  for (int j = 0; j < 9; ++j) { s[j] = tanhf(f1[j] + f2); mx = fmaxf(mx, s[j]); }
  float den = 0.f, num = 0.f;
#pragma unroll
  for (int j = 0; j < 9; ++j) { const float e = expf(s[j] - mx); den += e; num += e * logit[j]; }
  if (lane == 0) out[bt] = num / den;
}

// ---------------------------------------------------------------------------
// Host launch
// ---------------------------------------------------------------------------
extern "C" void kernel_launch(void* const* d_in, const int* in_sizes, int n_in,
                              void* d_out, int out_size, void* d_ws, size_t ws_size,
                              hipStream_t stream) {
  const float* iq     = (const float*)d_in[0];
  const float* nq     = (const float*)d_in[1];
  const float* isk    = (const float*)d_in[2];
  const float* ia     = (const float*)d_in[3];
  const int*   hidx   = (const int*)  d_in[4];
  const float* W_next = (const float*)d_in[5];
  const float* b_next = (const float*)d_in[6];
  const float* W_lin  = (const float*)d_in[7];
  const float* b_lin  = (const float*)d_in[8];
  const float* W_ih   = (const float*)d_in[9];
  const float* W_hh   = (const float*)d_in[10];
  const float* b_ih   = (const float*)d_in[11];
  const float* b_hh   = (const float*)d_in[12];
  const float* att1_w = (const float*)d_in[13];
  const float* att1_b = (const float*)d_in[14];
  const float* att2_w = (const float*)d_in[15];
  const float* att2_b = (const float*)d_in[16];
  float* out = (float*)d_out;
  (void)in_sizes; (void)n_in; (void)out_size; (void)ws_size;

  // workspace carve-up (keep 256B alignment; all sizes are multiples of 256B)
  char* p = (char*)d_ws;
  float*  ntr   = (float*)p;  p += (size_t)BT * HH * 4;     // next_trans f32
  float*  xp    = (float*)p;  p += (size_t)BT * G4H * 4;    // xp (T,B,4H) f32
  float*  oser  = (float*)p;  p += (size_t)BT * HH * 4;     // output_series f32
  float*  gates = (float*)p;  p += (size_t)BB * G4H * 4;
  float*  cbuf  = (float*)p;  p += (size_t)BB * HH * 4;
  __bf16* nqb   = (__bf16*)p; p += (size_t)BT * EE * 2;     // nq bf16
  __bf16* fab   = (__bf16*)p; p += (size_t)BT * FAW * 2;    // concat bf16
  __bf16* xinb  = (__bf16*)p; p += (size_t)BT * HH * 2;     // x_in bf16
  __bf16* hb    = (__bf16*)p; p += (size_t)BB * HH * 2;
  __bf16* wnb   = (__bf16*)p; p += (size_t)HH * EE * 2;
  __bf16* wlb   = (__bf16*)p; p += (size_t)HH * FAW * 2;
  __bf16* wihb  = (__bf16*)p; p += (size_t)G4H * HH * 2;
  __bf16* whhb  = (__bf16*)p; p += (size_t)G4H * HH * 2;

  // 0) one-time bf16 conversion of activations & weights
  cvt_bf16<<<(BT * EE / 8 + 255) / 256, 256, 0, stream>>>(nq, nqb, BT * EE);
  pack_fa <<<(BT * FAW / 8 + 255) / 256, 256, 0, stream>>>(iq, ia, isk, fab);
  cvt_bf16<<<(HH * EE / 8 + 255) / 256, 256, 0, stream>>>(W_next, wnb, HH * EE);
  cvt_bf16<<<(HH * FAW / 8 + 255) / 256, 256, 0, stream>>>(W_lin, wlb, HH * FAW);
  cvt_bf16<<<(G4H * HH / 8 + 255) / 256, 256, 0, stream>>>(W_ih, wihb, G4H * HH);
  cvt_bf16<<<(G4H * HH / 8 + 255) / 256, 256, 0, stream>>>(W_hh, whhb, G4H * HH);

  // 1) next_trans = relu(nq @ W_next^T + b_next)      blocks = 398*4
  gemm_bf16_lds<1, 0, 0><<<(BT / 64) * (HH / 64), 256, 0, stream>>>(
      nqb, wnb, b_next, nullptr, ntr, BT, HH, EE, 0, 0);

  // 2) x_in(bf16) = fa @ W_lin^T + b_lin              blocks = 398*4, K=768
  gemm_bf16_lds<0, 0, 1><<<(BT / 64) * (HH / 64), 256, 0, stream>>>(
      fab, wlb, b_lin, nullptr, xinb, BT, HH, FAW, 0, 0);

  // 3) xp[t,b,:] = x_in @ W_ih^T + b_ih               blocks = 398*16
  gemm_bf16_lds<0, 1, 0><<<(BT / 64) * (G4H / 64), 256, 0, stream>>>(
      xinb, wihb, b_ih, nullptr, xp, BT, G4H, HH, TTT, BB);

  // 4) LSTM scan over T (sequential dependency chain)
  zero_hc<<<(BB * HH) / 256, 256, 0, stream>>>(hb, cbuf);
  for (int t = 0; t < TTT; ++t) {
    gemm_bf16_lds<0, 0, 0><<<(BB / 64) * (G4H / 64), 256, 0, stream>>>(
        hb, whhb, b_hh, xp + (size_t)t * BB * G4H, gates, BB, G4H, HH, 0, 0);
    lstm_pointwise<<<(BB * HH) / 256, 256, 0, stream>>>(gates, hb, cbuf, oser, t);
  }

  // 5) gather + 9-way attention + softmax + weighted sum
  attn_out<<<BT / 8, 256, 0, stream>>>(oser, ntr, hidx, att1_w, att1_b,
                                       att2_w, att2_b, out);
}
